// TransformerLanguageModel_32538672234837
// MI455X (gfx1250) — matrix operand order repaired and checked by reading, not compiled, not run on hardware
//
#include <hip/hip_runtime.h>
#include <hip/hip_bf16.h>

typedef __attribute__((ext_vector_type(16))) _Float16 v16h;
typedef __attribute__((ext_vector_type(8)))  float    v8f;
typedef __attribute__((ext_vector_type(4)))  unsigned int u32x4;
typedef __attribute__((ext_vector_type(8)))  int      i32x8;
typedef __attribute__((ext_vector_type(4)))  int      i32x4;

namespace {
constexpr int B_ = 2, T_ = 1024, E_ = 1024, H_ = 16, HD_ = 64, L_ = 8, V_ = 32000, FF_ = 4096;
constexpr int BT_ = B_ * T_;
constexpr float EPS_ = 1e-5f;

// GEMM tiling: block = 256 threads = 8 wave32; C tile 128(M) x 128(N); K step 32.
// Each wave: 2 M-strips x 4 N-tiles = 8 accumulators -> 8 WMMA per K step.
// Double-buffered LDS: TDM (A) + global loads (B) for tile i+1 overlap WMMA on tile i.
constexpr int BM = 128, BN = 128, BK = 32;
constexpr int AS = BK + 8;   // sA  row stride (halves); produced by TDM pad feature
constexpr int BS = BK + 8;   // sBt row stride (halves): sBt[n][k]
}

// ---------------------------------------------------------------------------
// x[b,t,:] = tok_emb[context[b,t],:] + pos_emb[t,:]
// ---------------------------------------------------------------------------
__global__ void embed_kernel(const int* __restrict__ ctx, const float* __restrict__ tok,
                             const float* __restrict__ pos, float* __restrict__ x) {
  int idx = blockIdx.x * blockDim.x + threadIdx.x;
  if (idx >= BT_ * E_) return;
  int bt = idx / E_, e = idx - bt * E_;
  int t = bt % T_;
  x[idx] = tok[(size_t)ctx[bt] * E_ + e] + pos[(size_t)t * E_ + e];
}

// ---------------------------------------------------------------------------
// Row layernorm (E=1024), f32 in -> f16 out (GEMM operand precision).
// ---------------------------------------------------------------------------
__global__ __launch_bounds__(256) void layernorm_kernel(
    const float* __restrict__ x, const float* __restrict__ g, const float* __restrict__ b,
    _Float16* __restrict__ out) {
  __shared__ float rs[256], rs2[256];
  const int row = blockIdx.x;
  const float* xr = x + (size_t)row * E_;
  float s = 0.f, s2 = 0.f;
  for (int c = threadIdx.x; c < E_; c += 256) { float v = xr[c]; s += v; s2 += v * v; }
  rs[threadIdx.x] = s; rs2[threadIdx.x] = s2;
  __syncthreads();
  for (int st = 128; st > 0; st >>= 1) {
    if (threadIdx.x < st) { rs[threadIdx.x] += rs[threadIdx.x + st]; rs2[threadIdx.x] += rs2[threadIdx.x + st]; }
    __syncthreads();
  }
  const float mean = rs[0] * (1.0f / E_);
  const float var  = rs2[0] * (1.0f / E_) - mean * mean;
  const float inv  = rsqrtf(var + EPS_);
  _Float16* orow = out + (size_t)row * E_;
  for (int c = threadIdx.x; c < E_; c += 256)
    orow[c] = (_Float16)((xr[c] - mean) * inv * g[c] + b[c]);
}

// ---------------------------------------------------------------------------
// Issue one TDM descriptor: load 2D f16 tile (BK x BM rows) A[m0.., k0..]
// into LDS at ldsAddr with 16B padding after each 64B row (-> stride AS).
// Caller must be wave-uniform-branched so exactly one wave issues it.
// ---------------------------------------------------------------------------
__device__ __forceinline__ void tdm_load_a_tile(const _Float16* A, int M, int K,
                                                int m0, int k0, unsigned int ldsAddr) {
  unsigned long long ga = (unsigned long long)(uintptr_t)(A + (size_t)m0 * K + k0);
  u32x4 g0;
  g0[0] = 1u;                                   // count=1, user descriptor
  g0[1] = ldsAddr;                              // lds_addr [63:32]
  g0[2] = (unsigned int)ga;                     // global_addr [95:64]
  g0[3] = (unsigned int)((ga >> 32) & 0x1FFFFFFu) | (2u << 30);  // addr[56:32] | type=2
  i32x8 g1;
  // data_size=1 (2B) | pad_enable | pad_interval=3 (after 16 DWORDs) | pad_amount=3 (4 DWORDs)
  g1[0] = (1 << 16) | (1 << 20) | (3 << 22) | (3 << 25);
  g1[1] = (int)((K & 0xFFFF) << 16);            // abar=0, tensor_dim0[15:0]
  g1[2] = (int)((unsigned)K >> 16) | (int)((M & 0xFFFF) << 16);   // dim0 hi | dim1 lo
  g1[3] = (int)(((unsigned)M >> 16) & 0xFFFFu) | (BK << 16);      // dim1 hi | tile_dim0
  g1[4] = BM;                                   // tile_dim1=128, tile_dim2=0
  g1[5] = K;                                    // tensor_dim0_stride[31:0]
  g1[6] = 0;                                    // stride0 hi | stride1 lo
  g1[7] = 0;
  i32x4 gz = {};
#if defined(__clang_major__) && (__clang_major__ >= 23)
  i32x8 gz8 = {};
  __builtin_amdgcn_tensor_load_to_lds(g0, g1, gz, gz, gz8, 0);
#else
  __builtin_amdgcn_tensor_load_to_lds(g0, g1, gz, gz, 0);
#endif
}

// ---------------------------------------------------------------------------
// WMMA GEMM: C[M,N] = A_f16[M,K] @ B_f32[K,N] (+bias) (+relu)
// mode 0: store f16   mode 1: store f32 (non-temporal)   mode 2: resid += result
// Software pipelined: while WMMAs consume buffer `cur`, the TDM moves the next
// A tile and the block stages the next B tile (f32->f16, transposed) into `nxt`.
// ---------------------------------------------------------------------------
__global__ __launch_bounds__(256) void gemm_wmma_kernel(
    const _Float16* __restrict__ A, const float* __restrict__ Bm,
    const float* __restrict__ bias,
    _Float16* __restrict__ outh, float* __restrict__ outf, float* __restrict__ resid,
    int M, int N, int K, int mode, int do_relu) {
  __shared__ _Float16 sA[2][BM * AS];    // 2 x (128 x 32, padded to 40 by TDM)
  __shared__ _Float16 sBt[2][BN * BS];   // 2 x (128(n) x 32(k), padded)

  const int tid  = threadIdx.x;
  const int lane = tid & 31;
  const int wv   = tid >> 5;
  const int lm   = lane & 15;
  const int grp  = lane >> 4;
  const int wtm  = wv & 3;            // wave's 16-row M sub-tile within each 64-row strip
  const int wtn  = wv >> 2;           // wave's 64-col N half (0..1)
  const int m0 = blockIdx.y * BM;
  const int n0 = blockIdx.x * BN;

  const unsigned int sA_lds[2] = {(unsigned int)(uintptr_t)(&sA[0][0]),
                                  (unsigned int)(uintptr_t)(&sA[1][0])};

  auto stageB = [&](int k0, int buf) {
#pragma unroll
    for (int i = 0; i < 4; ++i) {
      int cidx = tid + i * 256;       // 0..1023
      int krow = cidx >> 5;           // 0..31
      int cc   = (cidx & 31) * 4;     // 0..124
      const float4 vb = *(const float4*)(Bm + (size_t)(k0 + krow) * N + n0 + cc);
      sBt[buf][(cc + 0) * BS + krow] = (_Float16)vb.x;
      sBt[buf][(cc + 1) * BS + krow] = (_Float16)vb.y;
      sBt[buf][(cc + 2) * BS + krow] = (_Float16)vb.z;
      sBt[buf][(cc + 3) * BS + krow] = (_Float16)vb.w;
    }
  };

  v8f zero8 = {};
  v8f acc[2][4];
#pragma unroll
  for (int s = 0; s < 2; ++s)
#pragma unroll
    for (int t = 0; t < 4; ++t) acc[s][t] = zero8;

  // prologue: stage tile 0 into buffer 0
  if (wv == 0) tdm_load_a_tile(A, M, K, m0, 0, sA_lds[0]);
  stageB(0, 0);
  if (wv == 0) __builtin_amdgcn_s_wait_tensorcnt(0);
  __syncthreads();

  const int nk = K / BK;
  for (int it = 0; it < nk; ++it) {
    const int cur = it & 1, nxt = cur ^ 1;
    // stage tile it+1 into the alternate buffer; overlaps with WMMA below
    if (it + 1 < nk) {
      if (wv == 0) tdm_load_a_tile(A, M, K, m0, (it + 1) * BK, sA_lds[nxt]);
      stageB((it + 1) * BK, nxt);
      if (it + 2 < nk) __builtin_prefetch(Bm + (size_t)(it + 2) * BK * N + n0, 0, 1);
    }

    // A fragments (16x32 f16, contiguous k order, matching B) for the wave's 2 strips:
    // lane: m = lane%16, g = lane/16; vgpr v holds K = 16g + 2v, +1
    v16h af[2];
#pragma unroll
    for (int s = 0; s < 2; ++s) {
      const _Float16* ap = &sA[cur][(s * 64 + wtm * 16 + lm) * AS + 16 * grp];
#pragma unroll
      for (int vv = 0; vv < 8; ++vv) {
        int off = 2 * vv;
        af[s][2 * vv]     = ap[off];
        af[s][2 * vv + 1] = ap[off + 1];
      }
    }
    // 4 B fragments (32x16 each): lane: n = lane%16, Kbase = 16*(lane/16);
    // each B fragment is reused by both M strips.
#pragma unroll
    for (int t = 0; t < 4; ++t) {
      v16h bf;
      const _Float16* bp = &sBt[cur][(wtn * 64 + t * 16 + lm) * BS + grp * 16];
#pragma unroll
      for (int e = 0; e < 16; ++e) bf[e] = bp[e];
#pragma unroll
      for (int s = 0; s < 2; ++s)
        acc[s][t] = __builtin_amdgcn_wmma_f32_16x16x32_f16(false, af[s], false, bf,
                                                           (short)0, acc[s][t], false, false);
    }
    // drain this iteration's TDM (targets `nxt`) before the barrier releases it
    if (wv == 0) __builtin_amdgcn_s_wait_tensorcnt(0);
    __syncthreads();
  }

  // epilogue: C layout -> lane holds col n=lane%16, vgpr j -> row j + 8*(lane/16)
#pragma unroll
  for (int s = 0; s < 2; ++s) {
    const int row0 = m0 + s * 64 + wtm * 16 + 8 * grp;
    const int col0 = n0 + wtn * 64 + lm;
#pragma unroll
    for (int t = 0; t < 4; ++t) {
#pragma unroll
      for (int j = 0; j < 8; ++j) {
        int r = row0 + j;
        int c = col0 + t * 16;
        float val = acc[s][t][j];
        if (bias)    val += bias[c];
        if (do_relu) val  = fmaxf(val, 0.0f);
        size_t idx = (size_t)r * N + c;
        if (mode == 0)      outh[idx] = (_Float16)val;
        else if (mode == 1) __builtin_nontemporal_store(val, &outf[idx]);
        else                resid[idx] += val;
      }
    }
  }
}

// ---------------------------------------------------------------------------
// Flash-style causal attention. One wave per (b, h, 16-row q tile).
// Streams 32-key chunks: S = Q(16x64) Kᵀ via 4 wmma, online softmax,
// O += P(16x32) V(32x64) via 4 wmma. scale = 1/sqrt(E) per reference.
// ---------------------------------------------------------------------------
__global__ __launch_bounds__(32) void attn_kernel(
    const _Float16* __restrict__ q, const _Float16* __restrict__ k,
    const _Float16* __restrict__ v, _Float16* __restrict__ o, float scale) {
  constexpr int QS = 72, KS = 72, VS = 40, PS = 40;
  __shared__ _Float16 sQ[16 * QS];
  __shared__ _Float16 sK[32 * KS];    // [key][dim]
  __shared__ _Float16 sVt[64 * VS];   // [dim][key] (transposed for B frags)
  __shared__ _Float16 sP[16 * PS];    // probs, A-operand staging

  const int lane = threadIdx.x;
  const int lm = lane & 15, grp = lane >> 4;
  const int nq = T_ / 16;
  const int qt = blockIdx.x % nq;
  const int hh = (blockIdx.x / nq) % H_;
  const int bb = blockIdx.x / (nq * H_);

  // stage Q tile (16 x 64)
#pragma unroll
  for (int i = 0; i < 4; ++i) {
    int cidx = lane + i * 32;
    int r = cidx >> 3, coff = (cidx & 7) * 8;
    const float4 qv = *(const float4*)(q + (size_t)(bb * T_ + qt * 16 + r) * E_ + hh * HD_ + coff);
    *(float4*)&sQ[r * QS + coff] = qv;
  }
  __syncthreads();
  v16h qa[2];
  {
    const _Float16* base = &sQ[lm * QS + 16 * grp];
#pragma unroll
    for (int c = 0; c < 2; ++c)
#pragma unroll
      for (int vv = 0; vv < 8; ++vv) {
        int off = c * 32 + 2 * vv;
        qa[c][2 * vv]     = base[off];
        qa[c][2 * vv + 1] = base[off + 1];
      }
  }

  v8f zero8 = {};
  v8f Ot[4] = {zero8, zero8, zero8, zero8};
  float Mrow[8], Lrow[8];
#pragma unroll
  for (int j = 0; j < 8; ++j) { Mrow[j] = -INFINITY; Lrow[j] = 0.f; }

  const int kcmax = (qt * 16 + 15) / 32;   // causal early exit
  for (int kc = 0; kc <= kcmax; ++kc) {
    __syncthreads();
    // stage K chunk (32x64) and V chunk transposed (64x32)
#pragma unroll
    for (int i = 0; i < 8; ++i) {
      int cidx = lane + i * 32;
      int r = cidx >> 3, coff = (cidx & 7) * 8;
      size_t src = (size_t)(bb * T_ + kc * 32 + r) * E_ + hh * HD_ + coff;
      const float4 kv = *(const float4*)(k + src);
      *(float4*)&sK[r * KS + coff] = kv;
      const float4 vv4 = *(const float4*)(v + src);
      const _Float16* pv = (const _Float16*)&vv4;
#pragma unroll
      for (int j = 0; j < 8; ++j) sVt[(coff + j) * VS + r] = pv[j];
    }
    __syncthreads();

    // scores: 2 key tiles x 2 dim chunks
    v8f s[2] = {zero8, zero8};
#pragma unroll
    for (int t = 0; t < 2; ++t) {
#pragma unroll
      for (int c = 0; c < 2; ++c) {
        v16h kb;  // B frag: lane key = t*16 + lane%16, dims c*32 + 16*(lane/16)..
        const _Float16* kp = &sK[(t * 16 + lm) * KS + c * 32 + grp * 16];
#pragma unroll
        for (int e = 0; e < 16; ++e) kb[e] = kp[e];
        s[t] = __builtin_amdgcn_wmma_f32_16x16x32_f16(false, qa[c], false, kb,
                                                      (short)0, s[t], false, false);
      }
    }
    // scale + causal mask
#pragma unroll
    for (int t = 0; t < 2; ++t)
#pragma unroll
      for (int j = 0; j < 8; ++j) {
        int kg = kc * 32 + t * 16 + lm;
        int qg = qt * 16 + j + 8 * grp;
        float sv = (float)s[t][j] * scale;
        s[t][j] = (kg <= qg) ? sv : -INFINITY;
      }
    // online softmax: row max / rescale / row sum (rows live in 16-lane groups)
    float alpha[8];
#pragma unroll
    for (int j = 0; j < 8; ++j) {
      float mv = fmaxf((float)s[0][j], (float)s[1][j]);
#pragma unroll
      for (int d = 1; d < 16; d <<= 1) mv = fmaxf(mv, __shfl_xor(mv, d, 32));
      float Mn = fmaxf(Mrow[j], mv);
      alpha[j] = (Mrow[j] == -INFINITY) ? 0.f : __expf(Mrow[j] - Mn);
      Mrow[j] = Mn;
    }
#pragma unroll
    for (int t = 0; t < 2; ++t)
#pragma unroll
      for (int j = 0; j < 8; ++j) {
        float p = __expf((float)s[t][j] - Mrow[j]);
        s[t][j] = p;
        sP[(j + 8 * grp) * PS + t * 16 + lm] = (_Float16)p;
      }
#pragma unroll
    for (int j = 0; j < 8; ++j) {
      float sv = (float)s[0][j] + (float)s[1][j];
#pragma unroll
      for (int d = 1; d < 16; d <<= 1) sv += __shfl_xor(sv, d, 32);
      Lrow[j] = alpha[j] * Lrow[j] + sv;
    }
#pragma unroll
    for (int t = 0; t < 4; ++t)
#pragma unroll
      for (int j = 0; j < 8; ++j) Ot[t][j] = (float)Ot[t][j] * alpha[j];
    __syncthreads();

    // O += P(16x32) @ V(32x64)
    v16h pa;
    {
      const _Float16* base = &sP[lm * PS + 16 * grp];
#pragma unroll
      for (int vv = 0; vv < 8; ++vv) {
        int off = 2 * vv;
        pa[2 * vv]     = base[off];
        pa[2 * vv + 1] = base[off + 1];
      }
    }
#pragma unroll
    for (int t = 0; t < 4; ++t) {
      v16h vb;
      const _Float16* vp = &sVt[(t * 16 + lm) * VS + grp * 16];
#pragma unroll
      for (int e = 0; e < 16; ++e) vb[e] = vp[e];
      Ot[t] = __builtin_amdgcn_wmma_f32_16x16x32_f16(false, pa, false, vb,
                                                     (short)0, Ot[t], false, false);
    }
  }
  // normalize and write o (f16, [B*T, E] with col = h*64+dim)
#pragma unroll
  for (int j = 0; j < 8; ++j) {
    float inv = 1.0f / Lrow[j];
    int r = bb * T_ + qt * 16 + j + 8 * grp;
#pragma unroll
    for (int t = 0; t < 4; ++t)
      o[(size_t)r * E_ + hh * HD_ + t * 16 + lm] = (_Float16)((float)Ot[t][j] * inv);
  }
}

// ---------------------------------------------------------------------------
// Cross-entropy: block per row over V=32000 logits; mean accumulated via atomics.
// ---------------------------------------------------------------------------
__global__ void zero_kernel(float* p) { if (threadIdx.x == 0 && blockIdx.x == 0) *p = 0.f; }

__global__ __launch_bounds__(256) void loss_kernel(const float* __restrict__ logits,
                                                   const int* __restrict__ tgt,
                                                   float* __restrict__ loss) {
  __shared__ float red[256];
  const int row = blockIdx.x;
  const float* lr = logits + (size_t)row * V_;
  float mx = -INFINITY;
  for (int c = threadIdx.x; c < V_; c += 256) mx = fmaxf(mx, lr[c]);
  red[threadIdx.x] = mx; __syncthreads();
  for (int st = 128; st > 0; st >>= 1) {
    if (threadIdx.x < st) red[threadIdx.x] = fmaxf(red[threadIdx.x], red[threadIdx.x + st]);
    __syncthreads();
  }
  mx = red[0]; __syncthreads();
  float s = 0.f;
  for (int c = threadIdx.x; c < V_; c += 256) s += __expf(lr[c] - mx);
  red[threadIdx.x] = s; __syncthreads();
  for (int st = 128; st > 0; st >>= 1) {
    if (threadIdx.x < st) red[threadIdx.x] += red[threadIdx.x + st];
    __syncthreads();
  }
  if (threadIdx.x == 0) {
    float lp = lr[tgt[row]] - mx - __logf(red[0]);
    atomicAdd(loss, -lp * (1.0f / BT_));
  }
}

// ---------------------------------------------------------------------------
extern "C" void kernel_launch(void* const* d_in, const int* in_sizes, int n_in,
                              void* d_out, int out_size, void* d_ws, size_t ws_size,
                              hipStream_t stream) {
  (void)in_sizes; (void)n_in; (void)out_size; (void)ws_size;
  const int*   ctx  = (const int*)d_in[0];
  const int*   tgt  = (const int*)d_in[1];
  const float* tok  = (const float*)d_in[2];
  const float* pos  = (const float*)d_in[3];
  const float* Wq   = (const float*)d_in[4];
  const float* Wk   = (const float*)d_in[5];
  const float* Wv   = (const float*)d_in[6];
  const float* Wo   = (const float*)d_in[7];
  const float* bo   = (const float*)d_in[8];
  const float* ln1g = (const float*)d_in[9];
  const float* ln1b = (const float*)d_in[10];
  const float* ln2g = (const float*)d_in[11];
  const float* ln2b = (const float*)d_in[12];
  const float* W1   = (const float*)d_in[13];
  const float* b1   = (const float*)d_in[14];
  const float* W2   = (const float*)d_in[15];
  const float* b2   = (const float*)d_in[16];
  const float* lnfg = (const float*)d_in[17];
  const float* lnfb = (const float*)d_in[18];
  const float* Wlm  = (const float*)d_in[19];
  const float* blm  = (const float*)d_in[20];

  float* logits = (float*)d_out;
  float* loss   = logits + (size_t)BT_ * V_;

  char* ws = (char*)d_ws;
  size_t off = 0;
  float*    x     = (float*)(ws + off);    off += (size_t)BT_ * E_ * 4;   // residual stream
  _Float16* h16   = (_Float16*)(ws + off); off += (size_t)BT_ * E_ * 2;   // LN output
  _Float16* q16   = (_Float16*)(ws + off); off += (size_t)BT_ * E_ * 2;
  _Float16* k16   = (_Float16*)(ws + off); off += (size_t)BT_ * E_ * 2;
  _Float16* v16   = (_Float16*)(ws + off); off += (size_t)BT_ * E_ * 2;
  _Float16* o16   = (_Float16*)(ws + off); off += (size_t)BT_ * E_ * 2;
  _Float16* mid16 = (_Float16*)(ws + off); off += (size_t)BT_ * FF_ * 2;  // ~44MB total

  const float scale = 1.0f / 32.0f;   // 1/sqrt(E), per reference (not head dim)
  dim3 blk256(256), blk32(32);

  embed_kernel<<<(BT_ * E_ + 255) / 256, blk256, 0, stream>>>(ctx, tok, pos, x);

  for (int l = 0; l < L_; ++l) {
    layernorm_kernel<<<BT_, blk256, 0, stream>>>(x, ln1g + (size_t)l * E_, ln1b + (size_t)l * E_, h16);
    gemm_wmma_kernel<<<dim3(E_ / BN, BT_ / BM), blk256, 0, stream>>>(
        h16, Wq + (size_t)l * E_ * E_, nullptr, q16, nullptr, nullptr, BT_, E_, E_, 0, 0);
    gemm_wmma_kernel<<<dim3(E_ / BN, BT_ / BM), blk256, 0, stream>>>(
        h16, Wk + (size_t)l * E_ * E_, nullptr, k16, nullptr, nullptr, BT_, E_, E_, 0, 0);
    gemm_wmma_kernel<<<dim3(E_ / BN, BT_ / BM), blk256, 0, stream>>>(
        h16, Wv + (size_t)l * E_ * E_, nullptr, v16, nullptr, nullptr, BT_, E_, E_, 0, 0);
    attn_kernel<<<B_ * H_ * (T_ / 16), blk32, 0, stream>>>(q16, k16, v16, o16, scale);
    gemm_wmma_kernel<<<dim3(E_ / BN, BT_ / BM), blk256, 0, stream>>>(
        o16, Wo + (size_t)l * E_ * E_, bo + (size_t)l * E_, nullptr, nullptr, x, BT_, E_, E_, 2, 0);
    layernorm_kernel<<<BT_, blk256, 0, stream>>>(x, ln2g + (size_t)l * E_, ln2b + (size_t)l * E_, h16);
    gemm_wmma_kernel<<<dim3(FF_ / BN, BT_ / BM), blk256, 0, stream>>>(
        h16, W1 + (size_t)l * E_ * FF_, b1 + (size_t)l * FF_, mid16, nullptr, nullptr, BT_, FF_, E_, 0, 1);
    gemm_wmma_kernel<<<dim3(E_ / BN, BT_ / BM), blk256, 0, stream>>>(
        mid16, W2 + (size_t)l * FF_ * E_, b2 + (size_t)l * E_, nullptr, nullptr, x, BT_, E_, FF_, 2, 0);
  }

  layernorm_kernel<<<BT_, blk256, 0, stream>>>(x, lnfg, lnfb, h16);
  gemm_wmma_kernel<<<dim3(V_ / BN, BT_ / BM), blk256, 0, stream>>>(
      h16, Wlm, blm, nullptr, logits, nullptr, BT_, V_, E_, 1, 0);

  zero_kernel<<<1, 1, 0, stream>>>(loss);
  loss_kernel<<<BT_, blk256, 0, stream>>>(logits, tgt, loss);
}
